// DIN_21028159881325
// MI455X (gfx1250) — matrix operand order repaired and checked by reading, not compiled, not run on hardware
//
#include <hip/hip_runtime.h>
#include <hip/hip_bf16.h>

typedef __attribute__((ext_vector_type(16))) __bf16 v16bf;
typedef __attribute__((ext_vector_type(8)))  float  v8f;

#define B_   4096
#define H_   100
#define D_   128
#define DIN_ 384
#define D1_  1024
#define D2_  512

union Frag { uint4 q[2]; v16bf v; };

__device__ __forceinline__ v16bf load_frag(const __bf16* p0, const __bf16* p1) {
  Frag f;
  f.q[0] = *reinterpret_cast<const uint4*>(p0);
  f.q[1] = *reinterpret_cast<const uint4*>(p1);
  return f.v;
}

// ---------------------------------------------------------------------------
// Attention pooling + concat, writes X16 = [items_emb | his_pool | user_emb]
// in bf16. One block per batch row; 8 waves; wave-per-history-row dots.
// ---------------------------------------------------------------------------
__global__ void attn_concat(const int* __restrict__ items,
                            const int* __restrict__ users,
                            const int* __restrict__ hist,
                            const float* __restrict__ item_table,
                            const float* __restrict__ user_table,
                            __bf16* __restrict__ X16) {
  int b = blockIdx.x;
  int tid = threadIdx.x;
  int lane = tid & 31, wave = tid >> 5;
  __shared__ float uemb[D_];
  __shared__ float accs[8][D_];
  if (tid < D_) uemb[tid] = user_table[(size_t)users[b] * D_ + tid];
  __syncthreads();
  float4 uc = reinterpret_cast<const float4*>(uemb)[lane];
  float4 acc = make_float4(0.f, 0.f, 0.f, 0.f);
  for (int h = wave; h < H_; h += 8) {   // uniform per wave -> full-wave shuffles
    const float4* hrow = reinterpret_cast<const float4*>(
        user_table + (size_t)hist[(size_t)b * H_ + h] * D_);
    float4 hc = hrow[lane];
    float p = hc.x*uc.x + hc.y*uc.y + hc.z*uc.z + hc.w*uc.w;
#pragma unroll
    for (int off = 16; off; off >>= 1) p += __shfl_xor(p, off);
    acc.x += p * hc.x; acc.y += p * hc.y; acc.z += p * hc.z; acc.w += p * hc.w;
  }
  accs[wave][lane * 4 + 0] = acc.x;
  accs[wave][lane * 4 + 1] = acc.y;
  accs[wave][lane * 4 + 2] = acc.z;
  accs[wave][lane * 4 + 3] = acc.w;
  __syncthreads();
  if (tid < D_) {
    float s = 0.f;
#pragma unroll
    for (int w = 0; w < 8; w++) s += accs[w][tid];
    size_t base = (size_t)b * DIN_;
    X16[base + tid]          = (__bf16)item_table[(size_t)items[b] * D_ + tid];
    X16[base + D_ + tid]     = (__bf16)s;
    X16[base + 2 * D_ + tid] = (__bf16)uemb[tid];
  }
}

// ---------------------------------------------------------------------------
// Convert f32 W[K][N] -> bf16 WT[N][K] (transposed so WMMA B-fragments are
// two contiguous 16B loads per lane).
// ---------------------------------------------------------------------------
__global__ void cvt_T(const float* __restrict__ W, __bf16* __restrict__ WT,
                      int K, int N) {
  int i = blockIdx.x * 256 + threadIdx.x;
  if (i >= K * N) return;
  int k = i / N, n = i - k * N;
  WT[(size_t)n * K + k] = (__bf16)W[i];
}

// ---------------------------------------------------------------------------
// C[M,N] = A16[M,K] * BT16[N,K]^T + bias  (f32 out)
// 256 threads = 8 waves; block tile 32(M) x 64(N); one 16x16 WMMA tile/wave.
// Fragment layouts per CDNA5 ISA 7.12.2 (16-bit A 16x32, B 32x16, f32 C/D).
// ---------------------------------------------------------------------------
__global__ void gemm_bf16(const __bf16* __restrict__ A,
                          const __bf16* __restrict__ BT,
                          const float* __restrict__ bias,
                          float* __restrict__ C,
                          int N, int K) {
  int lane = threadIdx.x & 31;
  int wave = threadIdx.x >> 5;
  int l16 = lane & 15;
  int hl  = lane >> 4;
  int m0 = blockIdx.y * 32 + (wave & 1) * 16;
  int n0 = blockIdx.x * 64 + (wave >> 1) * 16;
  const __bf16* arow = A  + (size_t)(m0 + l16) * K;
  const __bf16* brow = BT + (size_t)(n0 + l16) * K;
  v8f c = {};
  for (int k = 0; k < K; k += 32) {
    // A 16x32 bf16: lanes0-15 K={0..7,16..23}, lanes16-31 K={8..15,24..31}
    v16bf a = load_frag(arow + k + hl * 8, arow + k + 16 + hl * 8);
    // B 32x16 bf16 (from transposed W): lanes0-15 K={0..7,16..23}, lanes16-31 K={8..15,24..31}
    v16bf b = load_frag(brow + k + hl * 8, brow + k + 16 + hl * 8);
    if (k + 32 < K) {                       // uniform branch: EXEC untouched
      __builtin_prefetch(arow + k + 32, 0, 3);
      __builtin_prefetch(brow + k + 32, 0, 3);
    }
    c = __builtin_amdgcn_wmma_f32_16x16x32_bf16(
        /*neg_a=*/false, a, /*neg_b=*/false, b,
        /*c_mod=*/(short)0, c, /*reuse_a=*/false, /*reuse_b=*/false);
  }
  int col = n0 + l16;
  float bv = bias[col];
  float* crow = C + (size_t)(m0 + hl * 8) * N + col;   // D layout: VGPR r -> M=r (+8 for hi lanes)
#pragma unroll
  for (int r = 0; r < 8; r++)
    crow[(size_t)r * N] = c[r] + bv;
}

// ---------------------------------------------------------------------------
// BatchNorm column stats: deterministic two-stage reduction (no fp atomics).
// ---------------------------------------------------------------------------
__global__ void bn_part(const float* __restrict__ Y, int cols, int rows_per_chunk,
                        float2* __restrict__ part) {
  int c = blockIdx.x * blockDim.x + threadIdx.x;
  int chunk = blockIdx.y;
  const float* p = Y + (size_t)chunk * rows_per_chunk * cols + c;
  float s = 0.f, s2 = 0.f;
  for (int r = 0; r < rows_per_chunk; r++) {
    float v = p[(size_t)r * cols];
    s += v; s2 += v * v;
  }
  part[(size_t)chunk * cols + c] = make_float2(s, s2);
}

__global__ void bn_fin(const float2* __restrict__ part, int cols, int nchunk,
                       float inv_rows, float* __restrict__ mu,
                       float* __restrict__ rstd) {
  int c = blockIdx.x * blockDim.x + threadIdx.x;
  float s = 0.f, s2 = 0.f;
  for (int k = 0; k < nchunk; k++) {
    float2 v = part[(size_t)k * cols + c];
    s += v.x; s2 += v.y;
  }
  float m = s * inv_rows;
  float var = s2 * inv_rows - m * m;   // biased, as nn.BatchNorm1d training stats
  mu[c] = m;
  rstd[c] = rsqrtf(var + 1e-5f);
}

// ---------------------------------------------------------------------------
// Fused BatchNorm-apply + Dice (per-row stats). One block per row.
// FINAL=false: emit bf16 for next GEMM.  FINAL=true: fuse x@W3+b3 -> out[b].
// ---------------------------------------------------------------------------
template <int COLS, bool FINAL>
__global__ void bn_dice(const float* __restrict__ Y,
                        const float* __restrict__ mu, const float* __restrict__ rstd,
                        const float* __restrict__ g, const float* __restrict__ be,
                        const float* __restrict__ alpha,
                        __bf16* __restrict__ Z16,
                        const float* __restrict__ W3, const float* __restrict__ b3,
                        float* __restrict__ out) {
  __shared__ float xn[COLS];
  __shared__ float red[256];
  int b = blockIdx.x, tid = threadIdx.x;
  const float* yrow = Y + (size_t)b * COLS;
  float a = alpha[0];
  float lsum = 0.f;
  for (int c = tid; c < COLS; c += 256) {
    float v = (yrow[c] - mu[c]) * rstd[c] * g[c] + be[c];
    xn[c] = v;
    lsum += v;
  }
  red[tid] = lsum; __syncthreads();
#pragma unroll
  for (int s = 128; s > 0; s >>= 1) { if (tid < s) red[tid] += red[tid + s]; __syncthreads(); }
  float avg = red[0] * (1.0f / COLS);
  __syncthreads();
  float lvar = 0.f;
  for (int c = tid; c < COLS; c += 256) {
    float d = xn[c] - avg;
    lvar += d * d + 0.001f;             // sum((x-avg)^2 + DICE_EPS)
  }
  red[tid] = lvar; __syncthreads();
#pragma unroll
  for (int s = 128; s > 0; s >>= 1) { if (tid < s) red[tid] += red[tid + s]; __syncthreads(); }
  float rs = rsqrtf(red[0]);
  __syncthreads();
  float lacc = 0.f;
  for (int c = tid; c < COLS; c += 256) {
    float x = xn[c];
    float ps = 1.0f / (1.0f + __expf(-(x - avg) * rs));
    float o = x * (ps + (1.0f - ps) * a);
    if (FINAL) lacc += o * W3[c];
    else       Z16[(size_t)b * COLS + c] = (__bf16)o;
  }
  if (FINAL) {
    red[tid] = lacc; __syncthreads();
#pragma unroll
    for (int s = 128; s > 0; s >>= 1) { if (tid < s) red[tid] += red[tid + s]; __syncthreads(); }
    if (tid == 0) out[b] = red[0] + b3[0];
  }
}

// ---------------------------------------------------------------------------
extern "C" void kernel_launch(void* const* d_in, const int* in_sizes, int n_in,
                              void* d_out, int out_size, void* d_ws, size_t ws_size,
                              hipStream_t stream) {
  const int*   items  = (const int*)d_in[0];
  const int*   users  = (const int*)d_in[1];
  const int*   hist   = (const int*)d_in[2];
  const float* item_t = (const float*)d_in[3];
  const float* user_t = (const float*)d_in[4];
  const float* W1  = (const float*)d_in[5];
  const float* b1  = (const float*)d_in[6];
  const float* g1  = (const float*)d_in[7];
  const float* be1 = (const float*)d_in[8];
  const float* a1  = (const float*)d_in[9];
  const float* W2  = (const float*)d_in[10];
  const float* b2  = (const float*)d_in[11];
  const float* g2  = (const float*)d_in[12];
  const float* be2 = (const float*)d_in[13];
  const float* a2  = (const float*)d_in[14];
  const float* W3  = (const float*)d_in[15];
  const float* b3  = (const float*)d_in[16];
  float* out = (float*)d_out;

  char* ws = (char*)d_ws;
  size_t off = 0;
  auto alloc = [&](size_t bytes) -> void* {
    void* p = ws + off;
    off = (off + bytes + 255) & ~(size_t)255;
    return p;
  };
  __bf16* X16  = (__bf16*)alloc((size_t)B_ * DIN_ * 2);
  __bf16* W1T  = (__bf16*)alloc((size_t)D1_ * DIN_ * 2);
  __bf16* W2T  = (__bf16*)alloc((size_t)D2_ * D1_ * 2);
  float*  Y    = (float*) alloc((size_t)B_ * D1_ * 4);   // reused for Y1 and Y2
  __bf16* Z16  = (__bf16*)alloc((size_t)B_ * D1_ * 2);
  float2* part = (float2*)alloc((size_t)32 * D1_ * sizeof(float2));
  float*  mu   = (float*) alloc((size_t)D1_ * 4);
  float*  rstd = (float*) alloc((size_t)D1_ * 4);

  // weight conversion (bf16, transposed)
  cvt_T<<<(DIN_ * D1_ + 255) / 256, 256, 0, stream>>>(W1, W1T, DIN_, D1_);
  cvt_T<<<(D1_ * D2_ + 255) / 256, 256, 0, stream>>>(W2, W2T, D1_, D2_);

  // gather + attention pooling + concat
  attn_concat<<<B_, 256, 0, stream>>>(items, users, hist, item_t, user_t, X16);

  // layer 1: GEMM -> BN stats -> BN+Dice (bf16 out)
  gemm_bf16<<<dim3(D1_ / 64, B_ / 32), 256, 0, stream>>>(X16, W1T, b1, Y, D1_, DIN_);
  bn_part<<<dim3(D1_ / 256, 32), 256, 0, stream>>>(Y, D1_, B_ / 32, part);
  bn_fin<<<D1_ / 256, 256, 0, stream>>>(part, D1_, 32, 1.0f / B_, mu, rstd);
  bn_dice<D1_, false><<<B_, 256, 0, stream>>>(Y, mu, rstd, g1, be1, a1, Z16,
                                              nullptr, nullptr, nullptr);

  // layer 2: GEMM -> BN stats -> BN+Dice fused with x@W3+b3 -> out
  gemm_bf16<<<dim3(D2_ / 64, B_ / 32), 256, 0, stream>>>(Z16, W2T, b2, Y, D2_, D1_);
  bn_part<<<dim3(D2_ / 256, 32), 256, 0, stream>>>(Y, D2_, B_ / 32, part);
  bn_fin<<<D2_ / 256, 256, 0, stream>>>(part, D2_, 32, 1.0f / B_, mu, rstd);
  bn_dice<D2_, true><<<B_, 256, 0, stream>>>(Y, mu, rstd, g2, be2, a2, nullptr,
                                             W3, b3, out);
}
